// lstm1_39376260170352
// MI455X (gfx1250) — matrix operand order repaired and checked by reading, not compiled, not run on hardware
//
#include <hip/hip_runtime.h>
#include <stdint.h>

// ---------------- problem constants ----------------
constexpr int BATCH = 64;
constexpr int SEQ   = 512;
constexpr int NINP  = 1024;
constexpr int NHID  = 1024;
constexpr int G4    = 4 * NHID;        // 4096
constexpr int BT    = BATCH * SEQ;     // 32768

// Gate columns are INTERLEAVED throughout: packed column n = j*4 + g,
// g in {0:i, 1:f, 2:o, 3:c}. A 16-wide WMMA N-tile therefore carries
// complete gate quadruples, letting the LSTM cell update fuse into the
// GEMM epilogue via LDS.

// ---------------- vector types ----------------
typedef __attribute__((ext_vector_type(16))) __bf16 v16bf;
typedef __attribute__((ext_vector_type(8)))  float  v8f;
typedef __attribute__((ext_vector_type(4)))  unsigned int u32x4;
typedef __attribute__((ext_vector_type(8)))  int          i32x8;
typedef __attribute__((ext_vector_type(4)))  int          i32x4;

union Frag {
    v16bf v;
    uint4 q[2];
};

// ---------------- workspace layout (bytes) ----------------
constexpr size_t OFF_XBF  = 0;                                    // [BT, NINP] bf16   64 MB
constexpr size_t OFF_WXBF = OFF_XBF  + (size_t)BT * NINP * 2;     // [NINP, G4] bf16    8 MB
constexpr size_t OFF_WHBF = OFF_WXBF + (size_t)NINP * G4 * 2;     // [NHID, G4] bf16    8 MB
constexpr size_t OFF_BIAS = OFF_WHBF + (size_t)NHID * G4 * 2;     // [G4] f32          16 KB
constexpr size_t OFF_C    = OFF_BIAS + (size_t)G4 * 4;            // [B, H] f32       256 KB
constexpr size_t OFF_HBF0 = OFF_C    + (size_t)BATCH * NHID * 4;  // [B, H] bf16      128 KB
constexpr size_t OFF_HBF1 = OFF_HBF0 + (size_t)BATCH * NHID * 2;  // [B, H] bf16      128 KB
constexpr size_t OFF_XG   = OFF_HBF1 + (size_t)BATCH * NHID * 2;  // [BT, G4] f32     512 MB

// LDS partition
constexpr int LDS_A_BYTES  = BATCH * NHID * 2;          // 128 KB  (A tile, bf16)
constexpr int LDS_G_BYTES  = BATCH * 128 * 4;           //  32 KB  (G staging, f32)
constexpr int LDS_STEP     = LDS_A_BYTES + LDS_G_BYTES; // 160 KB
constexpr int LDS_XPROJ    = LDS_A_BYTES;               // 128 KB

#if __has_builtin(__builtin_amdgcn_tensor_load_to_lds)
#define HAVE_TDM 1
#else
#define HAVE_TDM 0
#endif

// ---------------- helpers ----------------
__device__ __forceinline__ uint16_t f2bf(float f) {
    uint32_t u = __float_as_uint(f);
    uint32_t r = (u + 0x7FFFu + ((u >> 16) & 1u)) >> 16;  // RNE
    return (uint16_t)r;
}

__device__ __forceinline__ float sigmoidf_(float x) {
    return 1.0f / (1.0f + __expf(-x));
}

#if HAVE_TDM
// One-shot TDM DMA of n8 x 8-byte elements from global memory into LDS.
// 1D descriptor per CDNA5 ISA:
//   group0: count=1 | lds_addr | global_addr | type=2
//   group1: data_size=3 (8B), tensor_dim0 = tile_dim0 = n8, tensor_dim1 = 1
// This toolchain's builtin takes 6 args (g0, g1, g2, g3, g4, cpol).
__device__ __forceinline__ void tdm_load_1d(uint32_t lds_off, const void* gsrc,
                                            uint32_t n8) {
    uint64_t ga = (uint64_t)(uintptr_t)gsrc;
    u32x4 g0;
    g0[0] = 1u;                                   // count=1, user descriptor
    g0[1] = lds_off;                              // lds_addr (bytes)
    g0[2] = (uint32_t)ga;                         // global_addr[31:0]
    g0[3] = (uint32_t)(ga >> 32) | 0x80000000u;   // global_addr[56:32] | type=2
    i32x8 g1;
    g1[0] = 0x00030000;                           // data_size=3 (8 bytes)
    g1[1] = (int)((n8 & 0xFFFFu) << 16);          // tensor_dim0[15:0]  -> bits[63:48]
    g1[2] = (int)(((n8 >> 16) & 0xFFFFu) | (1u << 16)); // tensor_dim0[31:16], tensor_dim1=1
    g1[3] = (int)((n8 & 0xFFFFu) << 16);          // tile_dim0 -> bits[127:112]
    g1[4] = 0;                                    // tile_dim1=0, tile_dim2=0
    g1[5] = (int)n8;                              // tensor_dim0_stride[31:0]
    g1[6] = 0;
    g1[7] = 0;
    i32x4 zz4 = {0, 0, 0, 0};
    i32x8 zz8 = {0, 0, 0, 0, 0, 0, 0, 0};
    __builtin_amdgcn_tensor_load_to_lds(g0, g1, zz4, zz4, zz8, 0);
#if __has_builtin(__builtin_amdgcn_s_wait_tensorcnt)
    __builtin_amdgcn_s_wait_tensorcnt((short)0);
#else
    asm volatile("s_wait_tensorcnt 0x0" ::: "memory");
#endif
}
#endif

// stage `bytes` (multiple of 4096) from gsrc into LDS at As; all 256 threads
__device__ __forceinline__ void stage_tile(void* As, const void* gsrc, int bytes,
                                           int tid) {
#if HAVE_TDM
    if (tid < 32) {  // single wave issues the DMA and waits TENSORcnt
        tdm_load_1d((uint32_t)(uintptr_t)As, gsrc, (uint32_t)(bytes >> 3));
    }
#else
    const uint4* src = (const uint4*)gsrc;
    uint4* dst = (uint4*)As;
    for (int i = tid; i < bytes / 16; i += 256) dst[i] = src[i];
#endif
    __syncthreads();
}

// ---------------- conversion kernels ----------------
__global__ void cvt_x_bf16(const float* __restrict__ src, uint16_t* __restrict__ dst, int n) {
    int i = blockIdx.x * blockDim.x + threadIdx.x;
    if (i < n) dst[i] = f2bf(src[i]);
}

// pack 4 gate matrices [1024,1024] (row-major (k,j)) into [K, 4096] bf16, n = j*4+g
__global__ void cvt_w_bf16(const float* __restrict__ g0, const float* __restrict__ g1,
                           const float* __restrict__ g2, const float* __restrict__ g3,
                           uint16_t* __restrict__ dst) {
    int idx = blockIdx.x * blockDim.x + threadIdx.x;   // K*G4 = 4M
    int k = idx >> 12;
    int n = idx & 4095;
    int g = n & 3;
    int j = n >> 2;
    const float* W = (g == 0) ? g0 : (g == 1) ? g1 : (g == 2) ? g2 : g3;
    dst[idx] = f2bf(W[(size_t)k * 1024 + j]);
}

__global__ void lstm_init(float* __restrict__ c, uint16_t* __restrict__ hbf,
                          float* __restrict__ bias,
                          const float* __restrict__ bi, const float* __restrict__ bf_,
                          const float* __restrict__ bo, const float* __restrict__ bc) {
    int idx = blockIdx.x * blockDim.x + threadIdx.x;   // 65536 threads
    if (idx < BATCH * NHID) {
        c[idx] = 0.0f;
        hbf[idx] = 0;
    }
    if (idx < G4) {
        int g = idx & 3, j = idx >> 2;
        const float* b = (g == 0) ? bi : (g == 1) ? bf_ : (g == 2) ? bo : bc;
        bias[idx] = b[j];
    }
}

// ---------------- input projection: Xg = Xbf @ Wxbf + bias ----------------
// block = 8 waves sharing one 64-row A tile staged in LDS; wave -> 64x16 tile
__global__ void lstm_xproj(const uint16_t* __restrict__ Xbf,
                           const uint16_t* __restrict__ Wxbf,
                           const float* __restrict__ bias,
                           float* __restrict__ Xg) {
    extern __shared__ char smem[];
    uint16_t* As = (uint16_t*)smem;                 // [64][1024] bf16

    const int tid  = threadIdx.x;
    const int lane = tid & 31;
    const int wave = tid >> 5;
    const int rowgrp = blockIdx.x >> 5;             // 0..511 (BT/64)
    const int ntile  = (blockIdx.x & 31) * 8 + wave;// 0..255 (G4/16)

    stage_tile(As, Xbf + (size_t)rowgrp * 64 * NINP, LDS_A_BYTES, tid);

    const int r  = lane & 15;
    const int hh = lane >> 4;
    const int n0 = ntile * 16;

    v8f z = {};
    v8f acc[4] = {z, z, z, z};

    for (int kt = 0; kt < 32; ++kt) {
        const int k0 = kt * 32;
        Frag bfr;   // B: lane = N column, 16 K values (table order)
        const uint16_t* bp = Wxbf + (size_t)(k0 + hh * 8) * G4 + n0 + r;
#pragma unroll
        for (int s = 0; s < 16; ++s)
            ((uint16_t*)&bfr)[s] = bp[(size_t)((s < 8) ? s : s + 8) * G4];
#pragma unroll
        for (int m = 0; m < 4; ++m) {
            const uint16_t* ap = As + (m * 16 + r) * NINP + k0 + hh * 8;
            Frag afr;
            afr.q[0] = *(const uint4*)(ap);
            afr.q[1] = *(const uint4*)(ap + 16);
            acc[m] = __builtin_amdgcn_wmma_f32_16x16x32_bf16(
                false, afr.v, false, bfr.v, (short)0, acc[m], false, false);
        }
    }

    const int n = n0 + r;
    const float bn = bias[n];
#pragma unroll
    for (int m = 0; m < 4; ++m) {
        const int rowbase = rowgrp * 64 + m * 16 + hh * 8;
#pragma unroll
        for (int e = 0; e < 8; ++e) {
            Xg[(size_t)(rowbase + e) * G4 + n] = acc[m][e] + bn;
        }
    }
}

// ---------------- fused recurrent step ----------------
// grid = 32 blocks x 256 threads. Block covers 128 packed columns = 32 hidden j.
__global__ void lstm_step(const uint16_t* __restrict__ hbf_in,
                          const uint16_t* __restrict__ Whbf,
                          const float* __restrict__ Xg,
                          float* __restrict__ c_state,
                          uint16_t* __restrict__ hbf_out,
                          float* __restrict__ out, int t) {
    extern __shared__ char smem[];
    uint16_t* As = (uint16_t*)smem;                       // 128 KB: h tile
    float*    Gs = (float*)(smem + LDS_A_BYTES);          //  32 KB: [64][128]

    const int tid  = threadIdx.x;
    const int lane = tid & 31;
    const int wave = tid >> 5;

    // Phase 0: stage full h matrix (64 x 1024 bf16) via TDM
    stage_tile(As, hbf_in, LDS_A_BYTES, tid);

    // Phase 1: WMMA over K=1024; each wave: 64x16 tile at ntile
    const int ntile = blockIdx.x * 8 + wave;    // 0..255
    const int r  = lane & 15;
    const int hh = lane >> 4;
    const int n0 = ntile * 16;

    v8f z = {};
    v8f acc[4] = {z, z, z, z};

    for (int kt = 0; kt < 32; ++kt) {
        const int k0 = kt * 32;
        Frag bfr;
        const uint16_t* bp = Whbf + (size_t)(k0 + hh * 8) * G4 + n0 + r;
#pragma unroll
        for (int s = 0; s < 16; ++s)
            ((uint16_t*)&bfr)[s] = bp[(size_t)((s < 8) ? s : s + 8) * G4];
#pragma unroll
        for (int m = 0; m < 4; ++m) {
            const uint16_t* ap = As + (m * 16 + r) * NHID + k0 + hh * 8;
            Frag afr;
            afr.q[0] = *(const uint4*)(ap);
            afr.q[1] = *(const uint4*)(ap + 16);
            acc[m] = __builtin_amdgcn_wmma_f32_16x16x32_bf16(
                false, afr.v, false, bfr.v, (short)0, acc[m], false, false);
        }
    }

    // epilogue: add Xg[:, t, :] and stage into LDS
    const int nloc = wave * 16 + r;                 // 0..127 within block
    const int ng   = blockIdx.x * 128 + nloc;       // global packed column
#pragma unroll
    for (int m = 0; m < 4; ++m) {
        const int rowbase = m * 16 + hh * 8;
#pragma unroll
        for (int e = 0; e < 8; ++e) {
            const int b = rowbase + e;
            Gs[b * 128 + nloc] =
                acc[m][e] + Xg[((size_t)b * SEQ + t) * G4 + ng];
        }
    }
    __syncthreads();

    // Phase 2: LSTM cell update. Block owns j in [blockIdx*32, +32), all b.
#pragma unroll
    for (int k = 0; k < 8; ++k) {
        const int s  = tid + k * 256;               // 0..2047
        const int jl = s & 31;
        const int b  = s >> 5;
        const float4 gq = *(const float4*)(Gs + b * 128 + jl * 4);
        const int j   = blockIdx.x * 32 + jl;
        const int idx = b * NHID + j;
        float it = sigmoidf_(gq.x);
        float ft = sigmoidf_(gq.y);
        float ot = sigmoidf_(gq.z);
        float ct = tanhf(gq.w);
        float c  = ft * c_state[idx] + it * ct;
        float h  = ot * tanhf(c);
        c_state[idx]  = c;
        hbf_out[idx]  = f2bf(h);
        out[((size_t)b * SEQ + t) * NHID + j] = h;
    }
}

// h_T is outputs[:, SEQ-1, :]; c_T from c_state
__global__ void lstm_final(const float* __restrict__ out, const float* __restrict__ c,
                           float* __restrict__ out_tail) {
    int idx = blockIdx.x * blockDim.x + threadIdx.x;   // 65536
    int b = idx >> 10;
    int j = idx & 1023;
    out_tail[idx] = out[((size_t)b * SEQ + (SEQ - 1)) * NHID + j];
    out_tail[BATCH * NHID + idx] = c[idx];
}

// ---------------- launcher ----------------
extern "C" void kernel_launch(void* const* d_in, const int* in_sizes, int n_in,
                              void* d_out, int out_size, void* d_ws, size_t ws_size,
                              hipStream_t stream) {
    (void)in_sizes; (void)n_in; (void)out_size; (void)ws_size;
    const float* X    = (const float*)d_in[0];
    const float* W_xi = (const float*)d_in[1];
    const float* W_hi = (const float*)d_in[2];
    const float* b_i  = (const float*)d_in[3];
    const float* W_xf = (const float*)d_in[4];
    const float* W_hf = (const float*)d_in[5];
    const float* b_f  = (const float*)d_in[6];
    const float* W_xo = (const float*)d_in[7];
    const float* W_ho = (const float*)d_in[8];
    const float* b_o  = (const float*)d_in[9];
    const float* W_xc = (const float*)d_in[10];
    const float* W_hc = (const float*)d_in[11];
    const float* b_c  = (const float*)d_in[12];

    char* ws = (char*)d_ws;
    uint16_t* Xbf   = (uint16_t*)(ws + OFF_XBF);
    uint16_t* Wxbf  = (uint16_t*)(ws + OFF_WXBF);
    uint16_t* Whbf  = (uint16_t*)(ws + OFF_WHBF);
    float*    bias  = (float*)(ws + OFF_BIAS);
    float*    c_st  = (float*)(ws + OFF_C);
    uint16_t* hbf0  = (uint16_t*)(ws + OFF_HBF0);
    uint16_t* hbf1  = (uint16_t*)(ws + OFF_HBF1);
    float*    Xg    = (float*)(ws + OFF_XG);

    float* out      = (float*)d_out;
    float* out_tail = out + (size_t)BATCH * SEQ * NHID;

    // allow >64KB dynamic LDS (WGP has 320 KB)
    (void)hipFuncSetAttribute((const void*)lstm_step,
                              hipFuncAttributeMaxDynamicSharedMemorySize, LDS_STEP);
    (void)hipFuncSetAttribute((const void*)lstm_xproj,
                              hipFuncAttributeMaxDynamicSharedMemorySize, LDS_XPROJ);

    // 1) conversions / init
    {
        int n = BT * NINP;
        cvt_x_bf16<<<(n + 255) / 256, 256, 0, stream>>>(X, Xbf, n);
    }
    {
        int n = NINP * G4;
        cvt_w_bf16<<<(n + 255) / 256, 256, 0, stream>>>(W_xi, W_xf, W_xo, W_xc, Wxbf);
        cvt_w_bf16<<<(n + 255) / 256, 256, 0, stream>>>(W_hi, W_hf, W_ho, W_hc, Whbf);
    }
    lstm_init<<<(BATCH * NHID + 255) / 256, 256, 0, stream>>>(
        c_st, hbf0, bias, b_i, b_f, b_o, b_c);

    // 2) input projection GEMM (16384 blocks x 8 waves, A tile via TDM -> LDS)
    lstm_xproj<<<16384, 256, LDS_XPROJ, stream>>>(Xbf, Wxbf, bias, Xg);

    // 3) sequential scan: one fused kernel per step, ping-pong h buffers
    for (int t = 0; t < SEQ; ++t) {
        const uint16_t* hin  = (t & 1) ? hbf1 : hbf0;
        uint16_t*       hout = (t & 1) ? hbf0 : hbf1;
        lstm_step<<<32, 256, LDS_STEP, stream>>>(
            hin, Whbf, Xg, c_st, hout, out, t);
    }

    // 4) final states
    lstm_final<<<(BATCH * NHID + 255) / 256, 256, 0, stream>>>(out, c_st, out_tail);
}
